// Model_46480136077521
// MI455X (gfx1250) — matrix operand order repaired and checked by reading, not compiled, not run on hardware
//
#include <hip/hip_runtime.h>

// ---------------------------------------------------------------------------
// CDNA5 / gfx1250 implementation of the Transolver-style physics-attention net.
// All GEMM-shaped work (convs-as-27-tap-GEMMs, MLPs, projections, slice
// reductions) goes through v_wmma_f32_16x16x32_bf16 (wave32, 16x16 tiles).
// ---------------------------------------------------------------------------

#define NTOK   32768
#define CCH    128
#define HEADS  8
#define DHD    16
#define GSL    32
#define NLAY   2

typedef __attribute__((ext_vector_type(8)))  __bf16 v8bf;
typedef __attribute__((ext_vector_type(16))) __bf16 v16bf;
typedef __attribute__((ext_vector_type(8)))  float  v8f;

// ---- WMMA fragment helpers (contiguous-K layouts, same for A and B, wave32) -------------------
// A (16x32 bf16): lane m=l&15; lanes 0-15 hold K {0..15}, lanes 16-31
// hold K {16..31}  -> one contiguous 16-element chunk per lane.
__device__ __forceinline__ v16bf load_a_frag(const __bf16* row, int l) {
    return *(const v16bf*)(row + ((l >> 4) << 4));
}
// B (32x16 bf16) from k-major storage W[n][k]: lane col n=l&15, 16 contiguous
// K starting at (l>=16 ? 16 : 0).  `rowk` = &W[n][kbase].
__device__ __forceinline__ v16bf load_b_frag(const __bf16* rowk, int l) {
    return *(const v16bf*)(rowk + ((l >> 4) << 4));
}
__device__ __forceinline__ v8f wmma_bf16(v16bf a, v16bf b, v8f c) {
    return __builtin_amdgcn_wmma_f32_16x16x32_bf16(false, a, false, b,
                                                   (short)0, c, false, false);
}
__device__ __forceinline__ float gelu_exact(float v) {
    return 0.5f * v * (1.f + erff(v * 0.70710678118f));
}

// ---- misc init: zero page + combined bias (pre_b2 + placeholder) ----------
__global__ void misc_init(float* zp, float* cbias, const float* pre_b2,
                          const float* placeholder) {
    int t = threadIdx.x;
    if (t < 256) zp[t] = 0.f;
    if (t < CCH) cbias[t] = pre_b2[t] + placeholder[t];
}

// ---- weight convert/transpose: fp32 [K][O] -> bf16 [O][K] -----------------
__global__ void wtrans(const float* in, __bf16* out, int K, int O) {
    int i = blockIdx.x * 256 + threadIdx.x;
    if (i >= K * O) return;
    int o = i / K, k = i % K;
    out[i] = (__bf16)in[k * O + o];
}
// conv weights fp32 [O=128][I=128][27] -> bf16 [27][O][I]
__global__ void ctrans(const float* in, __bf16* out) {
    int i = blockIdx.x * 256 + threadIdx.x;
    if (i >= 27 * CCH * CCH) return;
    int t = i / (CCH * CCH), rem = i % (CCH * CCH);
    int o = rem / CCH, ii = rem % CCH;
    out[i] = (__bf16)in[(o * CCH + ii) * 27 + t];
}

// ---- pre-MLP stage 1: gelu(h @ pre_w1 + b1), K=4, stored bf16 [N][256] ----
__global__ void pre_hidden(const float* x, const float* fx, const float* w1,
                           const float* b1, __bf16* hid) {
    int i = blockIdx.x * 256 + threadIdx.x;   // N*256 threads
    int n = i >> 8, j = i & 255;
    float v = x[n * 3] * w1[j] + x[n * 3 + 1] * w1[256 + j] +
              x[n * 3 + 2] * w1[512 + j] + fx[n] * w1[768 + j] + b1[j];
    hid[i] = (__bf16)gelu_exact(v);
}

// ---- generic WMMA GEMM: out = [gelu](A[M,K] @ W[O,K]^T + bias) [+res] -----
__global__ void gemm_wmma(const __bf16* A, const __bf16* W, const float* bias,
                          const float* residual, float* outF, __bf16* outB,
                          int K, int O, int doGelu) {
    int l  = threadIdx.x;
    int m0 = blockIdx.x * 16;
    int col = blockIdx.y * 16 + (l & 15);
    v8f acc = {};
    const __bf16* arow = A + (size_t)(m0 + (l & 15)) * K;
    const __bf16* wrow = W + (size_t)col * K;
    for (int kb = 0; kb < K; kb += 32) {
        v16bf a = load_a_frag(arow + kb, l);
        v16bf b = load_b_frag(wrow + kb, l);
        acc = wmma_bf16(a, b, acc);
    }
    float bs = bias ? bias[col] : 0.f;
    int rbase = m0 + ((l >> 4) << 3);
#pragma unroll
    for (int r = 0; r < 8; ++r) {
        int row = rbase + r;
        float v = acc[r] + bs;
        if (doGelu) v = gelu_exact(v);
        if (residual) v += residual[(size_t)row * O + col];
        if (outF) outF[(size_t)row * O + col] = v;
        if (outB) outB[(size_t)row * O + col] = (__bf16)v;
    }
}

// ---- LayerNorm: z fp32 [N][128] -> zn bf16 [N][128] (one wave per token) --
__global__ void ln_kernel(const float* z, const float* g, const float* b,
                          __bf16* zn) {
    int wid = threadIdx.x >> 5, lane = threadIdx.x & 31;
    int tok = blockIdx.x * 8 + wid;
    const float* row = z + (size_t)tok * CCH;
    float4 v = *(const float4*)(row + lane * 4);
    float s = v.x + v.y + v.z + v.w;
    for (int m = 16; m; m >>= 1) s += __shfl_xor(s, m, 32);
    float mu = s * (1.f / 128.f);
    float d0 = v.x - mu, d1 = v.y - mu, d2 = v.z - mu, d3 = v.w - mu;
    float q = d0 * d0 + d1 * d1 + d2 * d2 + d3 * d3;
    for (int m = 16; m; m >>= 1) q += __shfl_xor(q, m, 32);
    float inv = rsqrtf(q * (1.f / 128.f) + 1e-5f);
    int c = lane * 4;
    zn[(size_t)tok * CCH + c + 0] = (__bf16)(d0 * inv * g[c + 0] + b[c + 0]);
    zn[(size_t)tok * CCH + c + 1] = (__bf16)(d1 * inv * g[c + 1] + b[c + 1]);
    zn[(size_t)tok * CCH + c + 2] = (__bf16)(d2 * inv * g[c + 2] + b[c + 2]);
    zn[(size_t)tok * CCH + c + 3] = (__bf16)(d3 * inv * g[c + 3] + b[c + 3]);
}

// ---- dual 3D conv as 27 shifted WMMA GEMMs (shared A fragments) -----------
// zn bf16 [N][128]; weights bf16 [27][O][I]; outputs:
//   xmid fp32 [head][N][16]   (for slice logits)
//   fxT  bf16 [head*16+dh][N] (k-major B operand for tok GEMM)
__global__ void conv_wmma(const __bf16* zn, const __bf16* wx, const __bf16* wf,
                          const float* bx, const float* bf,
                          float* xmid, __bf16* fxT, const __bf16* zp) {
    int l  = threadIdx.x;
    int tt = blockIdx.x;        // token tile (16 tokens, contiguous in d)
    int oc = blockIdx.y;        // out-channel tile == head
    int m  = l & 15;
    int tok = tt * 16 + m;
    int d = tok & 31, w = (tok >> 5) & 31, h = tok >> 10;
    v8f accx = {}, accf = {};
    for (int t = 0; t < 27; ++t) {
        int k0 = t / 9 - 1, k1 = (t / 3) % 3 - 1, k2 = t % 3 - 1;
        int hh = h + k0, ww = w + k1, dd = d + k2;
        bool ok = ((unsigned)hh < 32u) & ((unsigned)ww < 32u) &
                  ((unsigned)dd < 32u);
        const __bf16* arow =
            ok ? (zn + (size_t)((hh * 32 + ww) * 32 + dd) * CCH) : zp;
        const __bf16* wxrow = wx + (size_t)(t * CCH + oc * 16 + (l & 15)) * CCH;
        const __bf16* wfrow = wf + (size_t)(t * CCH + oc * 16 + (l & 15)) * CCH;
#pragma unroll
        for (int kb = 0; kb < CCH; kb += 32) {
            v16bf a  = load_a_frag(arow + kb, l);
            v16bf b0 = load_b_frag(wxrow + kb, l);
            accx = wmma_bf16(a, b0, accx);
            v16bf b1 = load_b_frag(wfrow + kb, l);
            accf = wmma_bf16(a, b1, accf);
        }
    }
    int n = l & 15;
    int ch = oc * 16 + n;
    float bxv = bx[ch], bfv = bf[ch];
    int rbase = tt * 16 + ((l >> 4) << 3);
#pragma unroll
    for (int r = 0; r < 8; ++r)
        xmid[((size_t)oc * NTOK + rbase + r) * DHD + n] = accx[r] + bxv;
    v8bf pk;
#pragma unroll
    for (int r = 0; r < 8; ++r) pk[r] = (__bf16)(accf[r] + bfv);
    *(v8bf*)(fxT + (size_t)ch * NTOK + rbase) = pk;   // 8 consecutive tokens
}

// ---- slice logits + softmax; writes swt [h][n][g] and swtT [h][g][n] ------
__global__ void slice_softmax(const float* xmid, const float* sw,
                              const float* sb, const float* temp,
                              __bf16* swt, __bf16* swtT) {
    int i = blockIdx.x * 256 + threadIdx.x;   // head*NTOK + n
    int head = i >> 15, n = i & (NTOK - 1);
    float xm[DHD];
    const float* xr = xmid + (size_t)i * DHD;
#pragma unroll
    for (int c = 0; c < DHD; ++c) xm[c] = xr[c];
    float tinv = 1.f / temp[head];
    float lg[GSL], mx = -3.4e38f;
#pragma unroll
    for (int g = 0; g < GSL; ++g) {
        float s = sb[g];
#pragma unroll
        for (int c = 0; c < DHD; ++c) s += xm[c] * sw[c * GSL + g];
        s *= tinv;
        lg[g] = s;
        mx = fmaxf(mx, s);
    }
    float den = 0.f;
#pragma unroll
    for (int g = 0; g < GSL; ++g) { lg[g] = expf(lg[g] - mx); den += lg[g]; }
    float dinv = 1.f / den;
#pragma unroll
    for (int g = 0; g < GSL; ++g) {
        float p = lg[g] * dinv;
        swt[(size_t)i * GSL + g] = (__bf16)p;
        swtT[((size_t)head * GSL + g) * NTOK + n] = (__bf16)p;
    }
}

// ---- norm[h][g] = sum_n swt  (one wave per (h,g) pair) --------------------
__global__ void norm_kernel(const __bf16* swtT, float* nrm) {
    int wid = threadIdx.x >> 5, lane = threadIdx.x & 31;
    int pair = blockIdx.x * 8 + wid;     // 0..255
    const __bf16* p = swtT + (size_t)pair * NTOK;
    float s = 0.f;
    for (int j = lane; j < NTOK; j += 32) s += (float)p[j];
    for (int m = 16; m; m >>= 1) s += __shfl_xor(s, m, 32);
    if (lane == 0) nrm[pair] = s;
}

// ---- tok GEMM: Tok0[h][g][c] = sum_n swtT[h][g][n] * fxT[h*16+c][n] -------
__global__ void tok_gemm(const __bf16* swtT, const __bf16* fxT, float* Tok0) {
    int l = threadIdx.x;
    int head = blockIdx.x, mt = blockIdx.y;  // mt: g-tile (0,1)
    v8f acc = {};
    const __bf16* arow = swtT + ((size_t)head * GSL + mt * 16 + (l & 15)) * NTOK;
    const __bf16* brow = fxT + ((size_t)head * DHD + (l & 15)) * NTOK;
    for (int kb = 0; kb < NTOK; kb += 32) {
        v16bf a = load_a_frag(arow + kb, l);
        v16bf b = load_b_frag(brow + kb, l);
        acc = wmma_bf16(a, b, acc);
    }
    int g0 = mt * 16 + ((l >> 4) << 3);
    int c  = l & 15;
#pragma unroll
    for (int r = 0; r < 8; ++r)
        Tok0[((size_t)head * GSL + g0 + r) * DHD + c] = acc[r];
}

// ---- tiny attention core over G=32 slices per head (fp32, LDS) ------------
__global__ void attn_core(const float* Tok0, const float* nrm, const float* wq,
                          const float* wk, const float* wv, __bf16* otT) {
    __shared__ float ks[HEADS][GSL][DHD];
    __shared__ float vs[HEADS][GSL][DHD];
    int head = threadIdx.x >> 5, g = threadIdx.x & 31;
    float tok[DHD];
    float ninv = 1.f / (nrm[head * GSL + g] + 1e-5f);
    const float* tr = Tok0 + ((size_t)head * GSL + g) * DHD;
#pragma unroll
    for (int c = 0; c < DHD; ++c) tok[c] = tr[c] * ninv;
    float q[DHD];
#pragma unroll
    for (int c = 0; c < DHD; ++c) {
        float sq = 0.f, sk = 0.f, sv = 0.f;
#pragma unroll
        for (int d = 0; d < DHD; ++d) {
            sq += tok[d] * wq[d * DHD + c];
            sk += tok[d] * wk[d * DHD + c];
            sv += tok[d] * wv[d * DHD + c];
        }
        q[c] = sq;
        ks[head][g][c] = sk;
        vs[head][g][c] = sv;
    }
    __syncthreads();
    float lg[GSL], mx = -3.4e38f;
#pragma unroll
    for (int g2 = 0; g2 < GSL; ++g2) {
        float s = 0.f;
#pragma unroll
        for (int c = 0; c < DHD; ++c) s += q[c] * ks[head][g2][c];
        s *= 0.25f;                         // DH^-0.5
        lg[g2] = s;
        mx = fmaxf(mx, s);
    }
    float den = 0.f;
#pragma unroll
    for (int g2 = 0; g2 < GSL; ++g2) { lg[g2] = expf(lg[g2] - mx); den += lg[g2]; }
    float dinv = 1.f / den;
    float o[DHD];
#pragma unroll
    for (int c = 0; c < DHD; ++c) o[c] = 0.f;
#pragma unroll
    for (int g2 = 0; g2 < GSL; ++g2) {
        float p = lg[g2] * dinv;
#pragma unroll
        for (int c = 0; c < DHD; ++c) o[c] += p * vs[head][g2][c];
    }
#pragma unroll
    for (int c = 0; c < DHD; ++c)
        otT[((size_t)head * DHD + c) * GSL + g] = (__bf16)o[c];
}

// ---- de-slice: attn_out[n][h*16+c] = sum_g swt[h][n][g]*ot[h][g][c] -------
__global__ void deslice(const __bf16* swt, const __bf16* otT,
                        __bf16* attn_out) {
    int l = threadIdx.x;
    int tt = blockIdx.x, head = blockIdx.y;
    const __bf16* arow = swt + ((size_t)head * NTOK + tt * 16 + (l & 15)) * GSL;
    const __bf16* brow = otT + ((size_t)head * DHD + (l & 15)) * GSL;
    v16bf a = load_a_frag(arow, l);
    v16bf b = load_b_frag(brow, l);
    v8f acc = {};
    acc = wmma_bf16(a, b, acc);
    int n = l & 15, rbase = tt * 16 + ((l >> 4) << 3);
#pragma unroll
    for (int r = 0; r < 8; ++r)
        attn_out[((size_t)(rbase + r)) * CCH + head * DHD + n] = (__bf16)acc[r];
}

// ---- final LN + 128->1 projection (one wave per token) --------------------
__global__ void final_kernel(const float* z, const float* g, const float* b,
                             const float* ow, const float* ob, float* out) {
    int wid = threadIdx.x >> 5, lane = threadIdx.x & 31;
    int tok = blockIdx.x * 8 + wid;
    const float* row = z + (size_t)tok * CCH;
    float4 v = *(const float4*)(row + lane * 4);
    float s = v.x + v.y + v.z + v.w;
    for (int m = 16; m; m >>= 1) s += __shfl_xor(s, m, 32);
    float mu = s * (1.f / 128.f);
    float d0 = v.x - mu, d1 = v.y - mu, d2 = v.z - mu, d3 = v.w - mu;
    float q = d0 * d0 + d1 * d1 + d2 * d2 + d3 * d3;
    for (int m = 16; m; m >>= 1) q += __shfl_xor(q, m, 32);
    float inv = rsqrtf(q * (1.f / 128.f) + 1e-5f);
    int c = lane * 4;
    float y = (d0 * inv * g[c] + b[c]) * ow[c] +
              (d1 * inv * g[c + 1] + b[c + 1]) * ow[c + 1] +
              (d2 * inv * g[c + 2] + b[c + 2]) * ow[c + 2] +
              (d3 * inv * g[c + 3] + b[c + 3]) * ow[c + 3];
    for (int m = 16; m; m >>= 1) y += __shfl_xor(y, m, 32);
    if (lane == 0) out[tok] = y + ob[0];
}

// ---------------------------------------------------------------------------
extern "C" void kernel_launch(void* const* d_in, const int* in_sizes, int n_in,
                              void* d_out, int out_size, void* d_ws,
                              size_t ws_size, hipStream_t stream) {
    const float* x           = (const float*)d_in[0];
    const float* fx          = (const float*)d_in[1];
    const float* pre_w1      = (const float*)d_in[2];
    const float* pre_b1      = (const float*)d_in[3];
    const float* pre_w2      = (const float*)d_in[4];
    const float* pre_b2      = (const float*)d_in[5];
    const float* placeholder = (const float*)d_in[6];
    const float* ln1_g       = (const float*)d_in[7];
    const float* ln1_b       = (const float*)d_in[8];
    const float* convx_w     = (const float*)d_in[9];
    const float* convx_b     = (const float*)d_in[10];
    const float* convfx_w    = (const float*)d_in[11];
    const float* convfx_b    = (const float*)d_in[12];
    const float* slice_w     = (const float*)d_in[13];
    const float* slice_b     = (const float*)d_in[14];
    const float* temperature = (const float*)d_in[15];
    const float* wq          = (const float*)d_in[16];
    const float* wk          = (const float*)d_in[17];
    const float* wv          = (const float*)d_in[18];
    const float* wo          = (const float*)d_in[19];
    const float* bo          = (const float*)d_in[20];
    const float* ln2_g       = (const float*)d_in[21];
    const float* ln2_b       = (const float*)d_in[22];
    const float* mlp_w1      = (const float*)d_in[23];
    const float* mlp_b1      = (const float*)d_in[24];
    const float* mlp_w2      = (const float*)d_in[25];
    const float* mlp_b2      = (const float*)d_in[26];
    const float* ln3_g       = (const float*)d_in[27];
    const float* ln3_b       = (const float*)d_in[28];
    const float* out_w       = (const float*)d_in[29];
    const float* out_b       = (const float*)d_in[30];

    char* ws = (char*)d_ws;
    size_t off = 0;
    auto alloc = [&](size_t bytes) {
        size_t r = off;
        off = (off + bytes + 255) & ~(size_t)255;
        return r;
    };
    const size_t CWSZ = 27 * CCH * CCH;     // elements per conv-weight layer
    size_t zp_o   = alloc(1024);
    size_t cb_o   = alloc(CCH * 4);
    size_t nrm_o  = alloc(HEADS * GSL * 4);
    size_t tok0_o = alloc(HEADS * GSL * DHD * 4);
    size_t ott_o  = alloc(HEADS * DHD * GSL * 2);
    size_t w2t_o  = alloc(256 * CCH * 2);
    size_t m1t_o  = alloc(NLAY * CCH * CCH * 2);
    size_t m2t_o  = alloc(NLAY * CCH * CCH * 2);
    size_t wot_o  = alloc(NLAY * CCH * CCH * 2);
    size_t cxt_o  = alloc(NLAY * CWSZ * 2);
    size_t cft_o  = alloc(NLAY * CWSZ * 2);
    size_t xh_o   = alloc((size_t)NTOK * 256 * 2); // hidden0 bf16 / xmid f32
    size_t z_o    = alloc((size_t)NTOK * CCH * 4);
    size_t zn_o   = alloc((size_t)NTOK * CCH * 2);
    size_t fxt_o  = alloc((size_t)NTOK * CCH * 2);
    size_t swt_o  = alloc((size_t)HEADS * NTOK * GSL * 2);
    size_t swtt_o = alloc((size_t)HEADS * NTOK * GSL * 2);
    size_t ab_o   = alloc((size_t)NTOK * CCH * 2);  // attn_out / mlp hidden

    float*   zp_f   = (float*)(ws + zp_o);
    __bf16*  zp_bf  = (__bf16*)(ws + zp_o);
    float*   cbias  = (float*)(ws + cb_o);
    float*   nrm    = (float*)(ws + nrm_o);
    float*   Tok0   = (float*)(ws + tok0_o);
    __bf16*  otT    = (__bf16*)(ws + ott_o);
    __bf16*  w2T    = (__bf16*)(ws + w2t_o);
    __bf16*  m1T    = (__bf16*)(ws + m1t_o);
    __bf16*  m2T    = (__bf16*)(ws + m2t_o);
    __bf16*  woT    = (__bf16*)(ws + wot_o);
    __bf16*  cxT    = (__bf16*)(ws + cxt_o);
    __bf16*  cfT    = (__bf16*)(ws + cft_o);
    __bf16*  hid0   = (__bf16*)(ws + xh_o);
    float*   xmid   = (float*)(ws + xh_o);
    float*   z      = (float*)(ws + z_o);
    __bf16*  zn     = (__bf16*)(ws + zn_o);
    __bf16*  fxT    = (__bf16*)(ws + fxt_o);
    __bf16*  swt    = (__bf16*)(ws + swt_o);
    __bf16*  swtT   = (__bf16*)(ws + swtt_o);
    __bf16*  abuf   = (__bf16*)(ws + ab_o);

    // --- weight prep + init (deterministic each call) ---
    misc_init<<<1, 256, 0, stream>>>(zp_f, cbias, pre_b2, placeholder);
    wtrans<<<(256 * CCH + 255) / 256, 256, 0, stream>>>(pre_w2, w2T, 256, CCH);
    for (int l = 0; l < NLAY; ++l) {
        wtrans<<<64, 256, 0, stream>>>(mlp_w1 + l * CCH * CCH,
                                       m1T + l * CCH * CCH, CCH, CCH);
        wtrans<<<64, 256, 0, stream>>>(mlp_w2 + l * CCH * CCH,
                                       m2T + l * CCH * CCH, CCH, CCH);
        wtrans<<<64, 256, 0, stream>>>(wo + l * CCH * CCH,
                                       woT + l * CCH * CCH, CCH, CCH);
        ctrans<<<(27 * CCH * CCH + 255) / 256, 256, 0, stream>>>(
            convx_w + l * CWSZ, cxT + l * CWSZ);
        ctrans<<<(27 * CCH * CCH + 255) / 256, 256, 0, stream>>>(
            convfx_w + l * CWSZ, cfT + l * CWSZ);
    }

    // --- preprocessing MLP ---
    pre_hidden<<<NTOK, 256, 0, stream>>>(x, fx, pre_w1, pre_b1, hid0);
    gemm_wmma<<<dim3(NTOK / 16, CCH / 16), 32, 0, stream>>>(
        hid0, w2T, cbias, nullptr, z, nullptr, 256, CCH, 0);

    // --- transformer layers ---
    for (int l = 0; l < NLAY; ++l) {
        ln_kernel<<<NTOK / 8, 256, 0, stream>>>(z, ln1_g + l * CCH,
                                                ln1_b + l * CCH, zn);
        conv_wmma<<<dim3(NTOK / 16, HEADS), 32, 0, stream>>>(
            zn, cxT + l * CWSZ, cfT + l * CWSZ, convx_b + l * CCH,
            convfx_b + l * CCH, xmid, fxT, zp_bf);
        slice_softmax<<<(HEADS * NTOK) / 256, 256, 0, stream>>>(
            xmid, slice_w + l * DHD * GSL, slice_b + l * GSL,
            temperature + l * HEADS, swt, swtT);
        norm_kernel<<<32, 256, 0, stream>>>(swtT, nrm);
        tok_gemm<<<dim3(HEADS, 2), 32, 0, stream>>>(swtT, fxT, Tok0);
        attn_core<<<1, 256, 0, stream>>>(Tok0, nrm, wq + l * DHD * DHD,
                                         wk + l * DHD * DHD,
                                         wv + l * DHD * DHD, otT);
        deslice<<<dim3(NTOK / 16, HEADS), 32, 0, stream>>>(swt, otT, abuf);
        gemm_wmma<<<dim3(NTOK / 16, CCH / 16), 32, 0, stream>>>(
            abuf, woT + l * CCH * CCH, bo + l * CCH, z, z, nullptr, CCH, CCH, 0);
        ln_kernel<<<NTOK / 8, 256, 0, stream>>>(z, ln2_g + l * CCH,
                                                ln2_b + l * CCH, zn);
        gemm_wmma<<<dim3(NTOK / 16, CCH / 16), 32, 0, stream>>>(
            zn, m1T + l * CCH * CCH, mlp_b1 + l * CCH, nullptr, nullptr, abuf,
            CCH, CCH, 1);
        gemm_wmma<<<dim3(NTOK / 16, CCH / 16), 32, 0, stream>>>(
            abuf, m2T + l * CCH * CCH, mlp_b2 + l * CCH, z, z, nullptr, CCH,
            CCH, 0);
    }

    // --- final LN + projection ---
    final_kernel<<<NTOK / 8, 256, 0, stream>>>(z, ln3_g, ln3_b, out_w, out_b,
                                               (float*)d_out);
    (void)in_sizes; (void)n_in; (void)out_size; (void)ws_size;
}
